// MultiHeadAttention_34900904247826
// MI455X (gfx1250) — matrix operand order repaired and checked by reading, not compiled, not run on hardware
//
#include <hip/hip_runtime.h>

// ---------------------------------------------------------------------------
// Types
// ---------------------------------------------------------------------------
typedef __bf16 bf16_t;
typedef __bf16 v16bf __attribute__((ext_vector_type(16)));
typedef float  v8f   __attribute__((ext_vector_type(8)));
typedef unsigned int u32x4 __attribute__((ext_vector_type(4)));

union Frag {                 // one WMMA A/B operand: 16 bf16 per lane = 8 VGPRs
  v16bf v;
  u32x4 u[2];
};

static __device__ __forceinline__ v8f vzero8() {
  v8f z = {0.f, 0.f, 0.f, 0.f, 0.f, 0.f, 0.f, 0.f};
  return z;
}

static __device__ __forceinline__ v8f wmma_bf16(v16bf a, v16bf b, v8f c) {
  return __builtin_amdgcn_wmma_f32_16x16x32_bf16(
      /*neg_a=*/false, a, /*neg_b=*/false, b,
      /*c_mod=*/(short)0, c, /*reuse_a=*/false, /*reuse_b=*/false);
}

// CDNA5 async global->LDS copy (16 bytes), tracked by ASYNCcnt.
// LDS offset = low 32 bits of the generic pointer (shared aperture layout).
static __device__ __forceinline__ void async_copy_b128(const bf16_t* g, bf16_t* l) {
  unsigned lds = (unsigned)(unsigned long long)(uintptr_t)l;
  unsigned long long ga = (unsigned long long)(uintptr_t)g;
  asm volatile("global_load_async_to_lds_b128 %0, %1, off"
               :: "v"(lds), "v"(ga) : "memory");
}

static __device__ __forceinline__ void wait_async0() {
  asm volatile("s_wait_asynccnt 0x0" ::: "memory");
}

// raw v_exp_f32 (no subnormal guard; args are <= 0, underflow->0 is fine)
static __device__ __forceinline__ float fast_exp2(float x) {
  return __builtin_amdgcn_exp2f(x);
}

// 16-lane XOR butterfly via ds_swizzle (group-of-32 mode, imm = xor<<10 | and)
#define SWZ_XOR(x, mask)                                                   \
  __builtin_bit_cast(float, __builtin_amdgcn_ds_swizzle(                   \
      __builtin_bit_cast(int, (x)), (((mask) << 10) | 0x1f)))

#define BATCH 2
#define SEQ   2048
#define HIDN  1024
#define NHEAD 16
#define HDIM  64
#define MTOK  (BATCH * SEQ)   // 4096 tokens

// scores use exp2: fold 1/sqrt(HD) * log2(e) into Q at projection time
#define QSCALE (0.125f * 1.4426950408889634f)

// ---------------------------------------------------------------------------
// fp32 -> bf16 conversion
// ---------------------------------------------------------------------------
__global__ __launch_bounds__(256)
void cvt_f32_bf16_kernel(const float* __restrict__ in, bf16_t* __restrict__ out, int n) {
  int i = blockIdx.x * 256 + threadIdx.x;
  if (i < n) out[i] = (bf16_t)in[i];
}

// ---------------------------------------------------------------------------
// GEMM with bias: out = ((A[M][K] * W[N][K]^T) + bias) * oscale
// MODE 0: store bf16 row-major [M][N]
// MODE 1: store bf16 per-head transposed Vt[b][h][d][s]   (for V)
// MODE 2: store f32 row-major [M][N]                      (final output)
// Block tile 128x256, BK=32, 8 waves (2x4), wave tile 64x64 = 4x4 WMMA tiles.
// Double-buffered LDS fed by async global->LDS; one barrier per K-step.
// ---------------------------------------------------------------------------
template <int MODE>
__global__ __launch_bounds__(256)
void gemm_bias_kernel(const bf16_t* __restrict__ A,
                      const bf16_t* __restrict__ W,
                      const float*  __restrict__ bias,
                      void* __restrict__ outp,
                      float oscale, int M, int N, int Kd) {
  __shared__ bf16_t sA[2][128][32];   // 2 x 8 KB
  __shared__ bf16_t sB[2][256][32];   // 2 x 16 KB

  const int t    = threadIdx.x;
  const int wave = t >> 5;
  const int lane = t & 31;
  const int l16  = lane & 15;
  const int kg   = lane >> 4;      // 0 or 1 (half-wave group)
  const int wm   = wave >> 2;      // 0..1
  const int wn   = wave & 3;       // 0..3
  const int bm   = blockIdx.x * 128;
  const int bn   = blockIdx.y * 256;

  v8f acc[4][4];
#pragma unroll
  for (int i = 0; i < 4; ++i)
#pragma unroll
    for (int j = 0; j < 4; ++j) acc[i][j] = vzero8();

  // async staging: A tile 128x32 (2 ops/thread), B tile 256x32 (4 ops/thread)
  const int ar = t >> 1;            // 0..127
  const int ac = (t & 1) * 16;      // 0 / 16
  auto issue_tile = [&](int buf, int kk) {
    const bf16_t* ga = A + (size_t)(bm + ar) * Kd + kk + ac;
    bf16_t* da = &sA[buf][ar][ac];
    async_copy_b128(ga, da);
    async_copy_b128(ga + 8, da + 8);
    const bf16_t* gb = W + (size_t)(bn + t) * Kd + kk;
    bf16_t* db = &sB[buf][t][0];
    async_copy_b128(gb,      db);
    async_copy_b128(gb + 8,  db + 8);
    async_copy_b128(gb + 16, db + 16);
    async_copy_b128(gb + 24, db + 24);
  };

  const int nk = Kd / 32;
  issue_tile(0, 0);
  for (int kk = 0; kk < nk; ++kk) {
    const int cur = kk & 1;
    wait_async0();          // own tile-loads for buffer `cur` complete
    __syncthreads();        // everyone's complete; everyone done reading 1-cur
    if (kk + 1 < nk) issue_tile(cur ^ 1, (kk + 1) * 32);

    Frag af[4], bfr[4];
#pragma unroll
    for (int i = 0; i < 4; ++i) {
      const bf16_t* p = &sA[cur][wm * 64 + i * 16 + l16][0];
      af[i].u[0] = *(const u32x4*)(p + kg * 8);
      af[i].u[1] = *(const u32x4*)(p + 16 + kg * 8);
    }
#pragma unroll
    for (int j = 0; j < 4; ++j) {
      const bf16_t* p = &sB[cur][wn * 64 + j * 16 + l16][0];
      bfr[j].u[0] = *(const u32x4*)(p + kg * 8);
      bfr[j].u[1] = *(const u32x4*)(p + 16 + kg * 8);
    }
#pragma unroll
    for (int i = 0; i < 4; ++i)
#pragma unroll
      for (int j = 0; j < 4; ++j)
        acc[i][j] = wmma_bf16(af[i].v, bfr[j].v, acc[i][j]);
  }

  // Epilogue. D layout: lane holds column ncol, rows m0..m0+7 in acc[..][v].
#pragma unroll
  for (int j = 0; j < 4; ++j) {
    const int ncol = bn + wn * 64 + j * 16 + l16;
    const float bv = bias[ncol];
#pragma unroll
    for (int i = 0; i < 4; ++i) {
      const int m0 = bm + wm * 64 + i * 16 + kg * 8;
      if (MODE == 0) {
        bf16_t* out = (bf16_t*)outp;
#pragma unroll
        for (int v = 0; v < 8; ++v)
          out[(size_t)(m0 + v) * N + ncol] = (bf16_t)((acc[i][j][v] + bv) * oscale);
      } else if (MODE == 1) {
        union { u32x4 q; bf16_t e[8]; } pk;
#pragma unroll
        for (int v = 0; v < 8; ++v) pk.e[v] = (bf16_t)((acc[i][j][v] + bv) * oscale);
        const int bt = m0 >> 11;            // token / SEQ
        const int s0 = m0 & (SEQ - 1);      // 8-aligned -> 16B-aligned store
        const int hh = ncol >> 6;
        const int dd = ncol & (HDIM - 1);
        bf16_t* out = (bf16_t*)outp +
                      (((size_t)bt * NHEAD + hh) * HDIM + dd) * SEQ + s0;
        *(u32x4*)out = pk.q;
      } else {
        float* out = (float*)outp;
#pragma unroll
        for (int v = 0; v < 8; ++v)
          out[(size_t)(m0 + v) * N + ncol] = (acc[i][j][v] + bv) * oscale;
      }
    }
  }
}

// ---------------------------------------------------------------------------
// Flash attention: one block = 64 queries of one (batch, head); 4 waves,
// each wave owns a 16-query strip. Online softmax (base-2) over key tiles of
// 64, double-buffered async K/V staging, one barrier per key tile.
// Q pre-scaled by 0.125*log2(e). Vt layout: [B][H][HD][S].
// ---------------------------------------------------------------------------
__global__ __launch_bounds__(128)
void flash_attn_kernel(const bf16_t* __restrict__ Q,
                       const bf16_t* __restrict__ Kmat,
                       const bf16_t* __restrict__ Vt,
                       bf16_t* __restrict__ O) {
  __shared__ bf16_t sK[2][64][64];    // key tile, [key][d]     2 x 8 KB
  __shared__ bf16_t sV[2][64][64];    // value tile, [d][key]   2 x 8 KB
  __shared__ bf16_t sP[4][16][64];    // per-wave P relayout    8 KB

  const int t    = threadIdx.x;
  const int wave = t >> 5;
  const int lane = t & 31;
  const int l16  = lane & 15;
  const int kg   = lane >> 4;
  const int qb   = blockIdx.x;        // query block 0..31
  const int bh   = blockIdx.y;        // 0..31
  const int b    = bh >> 4;
  const int h    = bh & 15;

  // Q A-fragments held in registers for the whole kernel (K = HD = 64 -> 2 frags)
  Frag qf[2];
  {
    const bf16_t* qp =
        Q + (size_t)(b * SEQ + qb * 64 + wave * 16 + l16) * HIDN + h * HDIM;
    qf[0].u[0] = *(const u32x4*)(qp + kg * 8);
    qf[0].u[1] = *(const u32x4*)(qp + 16 + kg * 8);
    qf[1].u[0] = *(const u32x4*)(qp + 32 + kg * 8);
    qf[1].u[1] = *(const u32x4*)(qp + 48 + kg * 8);
  }

  float m_run[8], l_run[8];
  v8f oacc[4];
#pragma unroll
  for (int v = 0; v < 8; ++v) { m_run[v] = -__builtin_inff(); l_run[v] = 0.f; }
#pragma unroll
  for (int n = 0; n < 4; ++n) oacc[n] = vzero8();

  const int lr = t >> 1;              // 0..63
  const int lc = (t & 1) * 32;        // 0 or 32
  auto issue_tile = [&](int buf, int kt) {
    const bf16_t* gk =
        Kmat + (size_t)(b * SEQ + kt * 64 + lr) * HIDN + h * HDIM + lc;
    bf16_t* dk = &sK[buf][lr][lc];
    async_copy_b128(gk,      dk);
    async_copy_b128(gk + 8,  dk + 8);
    async_copy_b128(gk + 16, dk + 16);
    async_copy_b128(gk + 24, dk + 24);
    const bf16_t* gv =
        Vt + ((size_t)(b * NHEAD + h) * HDIM + lr) * SEQ + kt * 64 + lc;
    bf16_t* dv = &sV[buf][lr][lc];
    async_copy_b128(gv,      dv);
    async_copy_b128(gv + 8,  dv + 8);
    async_copy_b128(gv + 16, dv + 16);
    async_copy_b128(gv + 24, dv + 24);
  };

  issue_tile(0, 0);
  for (int kt = 0; kt < SEQ / 64; ++kt) {
    const int cur = kt & 1;
    wait_async0();
    __syncthreads();
    if (kt + 1 < SEQ / 64) issue_tile(cur ^ 1, kt + 1);

    // S' = Q' K^T  (already in log2-e, scaled domain)
    v8f sc[4];
#pragma unroll
    for (int n = 0; n < 4; ++n) {
      Frag k0, k1;
      const bf16_t* kp = &sK[cur][n * 16 + l16][0];
      k0.u[0] = *(const u32x4*)(kp + kg * 8);
      k0.u[1] = *(const u32x4*)(kp + 16 + kg * 8);
      k1.u[0] = *(const u32x4*)(kp + 32 + kg * 8);
      k1.u[1] = *(const u32x4*)(kp + 48 + kg * 8);
      v8f z = vzero8();
      z = wmma_bf16(qf[0].v, k0.v, z);
      z = wmma_bf16(qf[1].v, k1.v, z);
      sc[n] = z;
    }

    // online softmax in exp2 domain (row = 8*kg + v; reduce over 16-lane group)
    float rmax[8];
#pragma unroll
    for (int v = 0; v < 8; ++v) {
      float mx = fmaxf(fmaxf(sc[0][v], sc[1][v]), fmaxf(sc[2][v], sc[3][v]));
      mx = fmaxf(mx, SWZ_XOR(mx, 8));
      mx = fmaxf(mx, SWZ_XOR(mx, 4));
      mx = fmaxf(mx, SWZ_XOR(mx, 2));
      mx = fmaxf(mx, SWZ_XOR(mx, 1));
      rmax[v] = mx;
    }
#pragma unroll
    for (int v = 0; v < 8; ++v) {
      float mnew  = fmaxf(m_run[v], rmax[v]);
      float alpha = fast_exp2(m_run[v] - mnew);
      m_run[v] = mnew;
      float rs = 0.f;
#pragma unroll
      for (int n = 0; n < 4; ++n) {
        float p = fast_exp2(sc[n][v] - mnew);
        sc[n][v] = p;
        rs += p;
      }
      rs += SWZ_XOR(rs, 8);
      rs += SWZ_XOR(rs, 4);
      rs += SWZ_XOR(rs, 2);
      rs += SWZ_XOR(rs, 1);
      l_run[v] = l_run[v] * alpha + rs;
#pragma unroll
      for (int n = 0; n < 4; ++n) oacc[n][v] *= alpha;
    }

    // re-layout P (D layout) -> A layout via wave-private LDS
#pragma unroll
    for (int n = 0; n < 4; ++n)
#pragma unroll
      for (int v = 0; v < 8; ++v)
        sP[wave][kg * 8 + v][n * 16 + l16] = (bf16_t)sc[n][v];

    Frag pf[2];
    {
      const bf16_t* pp = &sP[wave][l16][0];
      pf[0].u[0] = *(const u32x4*)(pp + kg * 8);
      pf[0].u[1] = *(const u32x4*)(pp + 16 + kg * 8);
      pf[1].u[0] = *(const u32x4*)(pp + 32 + kg * 8);
      pf[1].u[1] = *(const u32x4*)(pp + 48 + kg * 8);
    }

    // O += P @ V   (4 d sub-tiles, K-dim = 64 keys = 2 WMMA steps)
#pragma unroll
    for (int n = 0; n < 4; ++n) {
      Frag v0, v1;
      const bf16_t* vp = &sV[cur][n * 16 + l16][0];
      v0.u[0] = *(const u32x4*)(vp + kg * 8);
      v0.u[1] = *(const u32x4*)(vp + 16 + kg * 8);
      v1.u[0] = *(const u32x4*)(vp + 32 + kg * 8);
      v1.u[1] = *(const u32x4*)(vp + 48 + kg * 8);
      oacc[n] = wmma_bf16(pf[0].v, v0.v, oacc[n]);
      oacc[n] = wmma_bf16(pf[1].v, v1.v, oacc[n]);
    }
  }

  // epilogue: normalize (v_rcp_f32) and store into O ([B*S][HIDN] layout)
  float inv_l[8];
#pragma unroll
  for (int v = 0; v < 8; ++v) inv_l[v] = __builtin_amdgcn_rcpf(l_run[v]);

  const int q0 = qb * 64 + wave * 16 + kg * 8;
#pragma unroll
  for (int n = 0; n < 4; ++n) {
    const int d = n * 16 + l16;
#pragma unroll
    for (int v = 0; v < 8; ++v) {
      float val = oacc[n][v] * inv_l[v];
      O[(size_t)(b * SEQ + q0 + v) * HIDN + h * HDIM + d] = (bf16_t)val;
    }
  }
}

// ---------------------------------------------------------------------------
// Launch
// ---------------------------------------------------------------------------
extern "C" void kernel_launch(void* const* d_in, const int* in_sizes, int n_in,
                              void* d_out, int out_size, void* d_ws, size_t ws_size,
                              hipStream_t stream) {
  (void)in_sizes; (void)n_in; (void)out_size; (void)ws_size;

  const float* X  = (const float*)d_in[0];
  const float* Wq = (const float*)d_in[1];
  const float* bq = (const float*)d_in[2];
  const float* Wk = (const float*)d_in[3];
  const float* bk = (const float*)d_in[4];
  const float* Wv = (const float*)d_in[5];
  const float* bv = (const float*)d_in[6];
  const float* Wo = (const float*)d_in[7];
  const float* bo = (const float*)d_in[8];

  const size_t NX = (size_t)MTOK * HIDN;   // 4,194,304 elems
  const size_t NW = (size_t)HIDN * HIDN;   // 1,048,576 elems

  // workspace carve-out (bf16): X, 4 weights, Q, K, Vt, O  -> 48 MiB total
  bf16_t* Xb  = (bf16_t*)d_ws;
  bf16_t* Wqb = Xb  + NX;
  bf16_t* Wkb = Wqb + NW;
  bf16_t* Wvb = Wkb + NW;
  bf16_t* Wob = Wvb + NW;
  bf16_t* Qb  = Wob + NW;
  bf16_t* Kb  = Qb  + NX;
  bf16_t* Vtb = Kb  + NX;
  bf16_t* Ob  = Vtb + NX;

  cvt_f32_bf16_kernel<<<(int)((NX + 255) / 256), 256, 0, stream>>>(X,  Xb,  (int)NX);
  cvt_f32_bf16_kernel<<<(int)((NW + 255) / 256), 256, 0, stream>>>(Wq, Wqb, (int)NW);
  cvt_f32_bf16_kernel<<<(int)((NW + 255) / 256), 256, 0, stream>>>(Wk, Wkb, (int)NW);
  cvt_f32_bf16_kernel<<<(int)((NW + 255) / 256), 256, 0, stream>>>(Wv, Wvb, (int)NW);
  cvt_f32_bf16_kernel<<<(int)((NW + 255) / 256), 256, 0, stream>>>(Wo, Wob, (int)NW);

  dim3 gg(MTOK / 128, HIDN / 256);   // (32, 4)
  gemm_bias_kernel<0><<<gg, 256, 0, stream>>>(Xb, Wqb, bq, Qb,  QSCALE, MTOK, HIDN, HIDN);
  gemm_bias_kernel<0><<<gg, 256, 0, stream>>>(Xb, Wkb, bk, Kb,  1.0f,   MTOK, HIDN, HIDN);
  gemm_bias_kernel<1><<<gg, 256, 0, stream>>>(Xb, Wvb, bv, Vtb, 1.0f,   MTOK, HIDN, HIDN);

  flash_attn_kernel<<<dim3(SEQ / 64, BATCH * NHEAD), 128, 0, stream>>>(Qb, Kb, Vtb, Ob);

  gemm_bias_kernel<2><<<gg, 256, 0, stream>>>(Ob, Wob, bo, d_out, 1.0f, MTOK, HIDN, HIDN);
}
